// H_Encoder_28003186770536
// MI455X (gfx1250) — matrix operand order repaired and checked by reading, not compiled, not run on hardware
//
#include <hip/hip_runtime.h>

// ---------------------------------------------------------------------------
// CDNA5 (gfx1250, wave32) implementation of the hierarchical point encoder.
// All K%32==0 matmuls run on v_wmma_f32_16x16x32_bf16 (bf16 in, f32 acc),
// each wave register-blocked to a 32x32 output region (2x2 WMMA tiles) so
// every A/B fragment load feeds two WMMAs. Residual state, LN/BN stats,
// softmax are f32. Attention stages K/V in LDS via async global->LDS loads
// (GLOBAL_LOAD_ASYNC_TO_LDS_B128, ASYNCcnt) when the builtins are available.
// ---------------------------------------------------------------------------

typedef __attribute__((ext_vector_type(16))) __bf16 v16bf;
typedef __attribute__((ext_vector_type(8)))  float  v8f;
typedef int v4i __attribute__((vector_size(16)));   // async-LDS builtin ptr type

#define HEADS 6
#define NMASK 51

#if defined(__HIP_DEVICE_COMPILE__) && defined(__gfx1250__) && \
    __has_builtin(__builtin_amdgcn_global_load_async_to_lds_b128) && \
    __has_builtin(__builtin_amdgcn_s_wait_asynccnt)
#define HAVE_ASYNC_LDS 1
#else
#define HAVE_ASYNC_LDS 0
#endif

__device__ __forceinline__ unsigned short f2bf(float f) {
  unsigned u = __float_as_uint(f);
  unsigned r = u + 0x7FFFu + ((u >> 16) & 1u);   // round-to-nearest-even
  return (unsigned short)(r >> 16);
}
__device__ __forceinline__ float bf2f(unsigned short h) {
  return __uint_as_float(((unsigned)h) << 16);
}

union FragBF { v16bf v; uint4 q[2]; };

enum { F_ACC = 1, F_RELU = 2, F_GELU = 4 };

__device__ __forceinline__ v8f acc_init(const unsigned short* outB, long mbase,
                                        int hh, int col, long M, int N)
{
  v8f c;
  #pragma unroll
  for (int r = 0; r < 8; ++r) {
    long orow = mbase + hh * 8 + r;
    c[r] = (orow < M) ? bf2f(outB[orow * (long)N + col]) : 0.f;
  }
  return c;
}

__device__ __forceinline__ void epilogue(v8f c, long mbase, int hh, int col,
                                         bool cok, long M, int N,
                                         const float* bias, const float* resid,
                                         float* outF, unsigned short* outB,
                                         int flags)
{
  if (!cok) return;
  float bc = bias ? bias[col] : 0.f;
  #pragma unroll
  for (int r = 0; r < 8; ++r) {
    long orow = mbase + hh * 8 + r;      // C/D layout: vgpr r -> M = r + hh*8
    if (orow >= M) continue;
    float v = c[r] + bc;
    if (flags & F_RELU) v = fmaxf(v, 0.f);
    if (flags & F_GELU) v = 0.5f * v * (1.f + erff(v * 0.70710678f));
    long oi = orow * (long)N + col;
    if (resid) v += resid[oi];
    if (outF) outF[oi] = v;
    if (outB) outB[oi] = f2bf(v);
  }
}

// ---------------------------------------------------------------------------
// WMMA GEMM: C(MxN, f32 acc) = A(MxK bf16) @ W(NxK bf16, row stride ldw)^T
//   + bias + optional residual; optional relu/gelu; writes f32 and/or bf16.
// arow_div: logical A row -> physical A row / arow_div (row broadcast).
// F_ACC: seed accumulator from existing bf16 outB contents (split-K/concat).
// One wave -> 32x32 output (2x2 tiles, 4 accumulators): each A fragment is
// reused by 2 WMMAs and each B fragment by 2 WMMAs. K must be K%32==0.
// ---------------------------------------------------------------------------
__global__ __launch_bounds__(128) void k_gemm_bf16(
    const unsigned short* __restrict__ A, const unsigned short* __restrict__ W,
    int ldw, const float* __restrict__ bias, const float* __restrict__ resid,
    float* __restrict__ outF, unsigned short* __restrict__ outB,
    long M, int N, int K, int arow_div, int flags)
{
  int wid  = threadIdx.x >> 5;
  int lane = threadIdx.x & 31;
  long mt  = blockIdx.x;                 // 32-row tile
  int  nt  = blockIdx.y * 4 + wid;       // 32-col tile
  if (nt * 32 >= N) return;
  int m = lane & 15, hh = lane >> 4;
  long row0 = mt * 32 + m;
  long row1 = row0 + 16;
  long lr0 = (row0 < M ? row0 : M - 1) / arow_div;
  long lr1 = (row1 < M ? row1 : M - 1) / arow_div;
  const unsigned short* ar0 = A + lr0 * (long)K;
  const unsigned short* ar1 = A + lr1 * (long)K;
  int col0 = nt * 32 + m;
  bool n1ok = (nt * 32 + 16) < N;
  int col1 = n1ok ? (col0 + 16) : col0;
  const unsigned short* wr0 = W + (long)col0 * (long)ldw;
  const unsigned short* wr1 = W + (long)col1 * (long)ldw;

  v8f c00, c01, c10, c11;
  #pragma unroll
  for (int r = 0; r < 8; ++r) { c00[r] = 0.f; c01[r] = 0.f; c10[r] = 0.f; c11[r] = 0.f; }
  if (flags & F_ACC) {
    c00 = acc_init(outB, mt * 32,      hh, col0,      M, N);
    c10 = acc_init(outB, mt * 32 + 16, hh, col0,      M, N);
    if (n1ok) {
      c01 = acc_init(outB, mt * 32,      hh, col0 + 16, M, N);
      c11 = acc_init(outB, mt * 32 + 16, hh, col0 + 16, M, N);
    }
  }
  for (int k0 = 0; k0 < K; k0 += 32) {
    FragBF a0, a1, b0, b1;
    // A 16x32 frag: lane(m,hh): elems 0..7 = K[h*8..], 8..15 = K[16+h*8..]
    a0.q[0] = *(const uint4*)(ar0 + k0 + hh * 8);
    a0.q[1] = *(const uint4*)(ar0 + k0 + 16 + hh * 8);
    a1.q[0] = *(const uint4*)(ar1 + k0 + hh * 8);
    a1.q[1] = *(const uint4*)(ar1 + k0 + 16 + hh * 8);
    // B 32x16 frag: lane(n,hh): elems 0..7 = K[h*8..], 8..15 = K[16+h*8..] of weight row n
    const uint4* w0 = (const uint4*)(wr0 + k0 + hh * 8);
    const uint4* w1 = (const uint4*)(wr1 + k0 + hh * 8);
    b0.q[0] = w0[0]; b0.q[1] = w0[2];
    b1.q[0] = w1[0]; b1.q[1] = w1[2];
    c00 = __builtin_amdgcn_wmma_f32_16x16x32_bf16(false, a0.v, false, b0.v, (short)0, c00, false, false);
    c01 = __builtin_amdgcn_wmma_f32_16x16x32_bf16(false, a0.v, false, b1.v, (short)0, c01, false, false);
    c10 = __builtin_amdgcn_wmma_f32_16x16x32_bf16(false, a1.v, false, b0.v, (short)0, c10, false, false);
    c11 = __builtin_amdgcn_wmma_f32_16x16x32_bf16(false, a1.v, false, b1.v, (short)0, c11, false, false);
  }
  epilogue(c00, mt * 32,      hh, col0,      true, M, N, bias, resid, outF, outB, flags);
  epilogue(c01, mt * 32,      hh, col0 + 16, n1ok, M, N, bias, resid, outF, outB, flags);
  epilogue(c10, mt * 32 + 16, hh, col0,      true, M, N, bias, resid, outF, outB, flags);
  epilogue(c11, mt * 32 + 16, hh, col0 + 16, n1ok, M, N, bias, resid, outF, outB, flags);
}

// --------------------------- attention (K/V in LDS) ------------------------
template<int DH>
__global__ __launch_bounds__(256) void k_attn(
    const float* __restrict__ qkv, const unsigned char* __restrict__ am,
    unsigned short* __restrict__ o, int n, int d)
{
  extern __shared__ float smem[];
  int bh = blockIdx.x;
  int b = bh / HEADS, h = bh % HEADS;
  float* Ks = smem;
  float* Vs = smem + (size_t)n * DH;
#if HAVE_ASYNC_LDS
  // async global->LDS (ASYNCcnt): 16B per lane; K/V rows are 16B aligned
  const int CH = DH / 4;
  for (int t = threadIdx.x; t < n * CH; t += blockDim.x) {
    int i = t / CH, j = t % CH;
    const float* gK = qkv + ((size_t)(b * n + i)) * 3 * d + d + h * DH + j * 4;
    const float* gV = gK + d;
    __builtin_amdgcn_global_load_async_to_lds_b128(
        (v4i*)gK, (v4i*)(Ks + (size_t)i * DH + j * 4), 0, 0);
    __builtin_amdgcn_global_load_async_to_lds_b128(
        (v4i*)gV, (v4i*)(Vs + (size_t)i * DH + j * 4), 0, 0);
  }
  __builtin_amdgcn_s_wait_asynccnt(0);
  __syncthreads();
#else
  for (int t = threadIdx.x; t < n * DH; t += blockDim.x) {
    int i = t / DH, j = t % DH;
    const float* row = qkv + ((size_t)(b * n + i)) * 3 * d + h * DH;
    Ks[t] = row[d + j];
    Vs[t] = row[2 * d + j];
  }
  __syncthreads();
#endif
  float scale = rsqrtf((float)DH);
  for (int i = threadIdx.x; i < n; i += blockDim.x) {
    const float* qr = qkv + ((size_t)(b * n + i)) * 3 * d + h * DH;
    float q[DH];
    #pragma unroll
    for (int j = 0; j < DH; ++j) q[j] = qr[j] * scale;
    const unsigned char* amr = am ? (am + ((size_t)b * n + i) * n) : nullptr;
    float mx = -1e30f;
    for (int k = 0; k < n; ++k) {
      const float* kr = Ks + (size_t)k * DH;
      float s = 0.f;
      #pragma unroll
      for (int j = 0; j < DH; ++j) s += q[j] * kr[j];
      if (amr && amr[k]) s = -1e9f;
      mx = fmaxf(mx, s);
    }
    float acc[DH];
    #pragma unroll
    for (int j = 0; j < DH; ++j) acc[j] = 0.f;
    float den = 0.f;
    for (int k = 0; k < n; ++k) {
      const float* kr = Ks + (size_t)k * DH;
      float s = 0.f;
      #pragma unroll
      for (int j = 0; j < DH; ++j) s += q[j] * kr[j];
      if (amr && amr[k]) s = -1e9f;
      float w = __expf(s - mx);
      den += w;
      const float* vr = Vs + (size_t)k * DH;
      #pragma unroll
      for (int j = 0; j < DH; ++j) acc[j] += w * vr[j];
    }
    float inv = 1.f / den;
    unsigned short* orow = o + ((size_t)(b * n + i)) * d + h * DH;
    #pragma unroll
    for (int j = 0; j < DH; ++j) orow[j] = f2bf(acc[j] * inv);
  }
}

// --------------------------- layernorm (1 warp/row) ------------------------
__global__ void k_ln(const float* __restrict__ X, const float* __restrict__ s,
                     const float* __restrict__ bb, unsigned short* outB,
                     float* outF, long rows, int d)
{
  long warp = (long)blockIdx.x * (blockDim.x >> 5) + (threadIdx.x >> 5);
  int lane = threadIdx.x & 31;
  if (warp >= rows) return;
  const float* xr = X + warp * (long)d;
  float s1 = 0.f, s2 = 0.f;
  for (int c = lane; c < d; c += 32) { float v = xr[c]; s1 += v; s2 += v * v; }
  for (int o = 16; o > 0; o >>= 1) {
    s1 += __shfl_down(s1, o, 32);
    s2 += __shfl_down(s2, o, 32);
  }
  s1 = __shfl(s1, 0, 32); s2 = __shfl(s2, 0, 32);
  float mean = s1 / d;
  float inv  = rsqrtf(s2 / d - mean * mean + 1e-5f);
  for (int c = lane; c < d; c += 32) {
    float v = (xr[c] - mean) * inv * s[c] + bb[c];
    if (outB) outB[warp * (long)d + c] = f2bf(v);
    if (outF) outF[warp * (long)d + c] = v;
  }
}

// --------------------------- batchnorm (global stats) ----------------------
__global__ void k_colstats(const unsigned short* __restrict__ X, long rows,
                           int cols, float* st)
{
  int c = blockIdx.x * blockDim.x + threadIdx.x;
  if (c >= cols) return;
  long per = (rows + gridDim.y - 1) / gridDim.y;
  long r0 = per * blockIdx.y;
  long r1 = r0 + per; if (r1 > rows) r1 = rows;
  float s = 0.f, s2 = 0.f;
  for (long r = r0; r < r1; ++r) {
    float v = bf2f(X[r * (long)cols + c]);
    s += v; s2 += v * v;
  }
  atomicAdd(&st[c], s);
  atomicAdd(&st[cols + c], s2);
}

__global__ void k_bn(unsigned short* X, long rows, int cols, const float* st,
                     const float* gs, const float* gb, int relu)
{
  long t = (long)blockIdx.x * blockDim.x + threadIdx.x;
  long total = rows * (long)cols;
  if (t >= total) return;
  int c = (int)(t % cols);
  float mean = st[c] / (float)rows;
  float var  = st[cols + c] / (float)rows - mean * mean;
  float v = (bf2f(X[t]) - mean) * rsqrtf(var + 1e-5f) * gs[c] + gb[c];
  if (relu) v = fmaxf(v, 0.f);
  X[t] = f2bf(v);
}

// --------------------------- small elementwise kernels ---------------------
__global__ void k_f2bf(const float* __restrict__ x, unsigned short* o, long n) {
  long t = (long)blockIdx.x * blockDim.x + threadIdx.x;
  if (t < n) o[t] = f2bf(x[t]);
}
__global__ void k_zero_f(float* p, long n) {
  long t = (long)blockIdx.x * blockDim.x + threadIdx.x;
  if (t < n) p[t] = 0.f;
}
__global__ void k_fill_b(unsigned char* p, unsigned char v, long n) {
  long t = (long)blockIdx.x * blockDim.x + threadIdx.x;
  if (t < n) p[t] = v;
}
__global__ void k_copy_b(const unsigned char* s, unsigned char* d, long n) {
  long t = (long)blockIdx.x * blockDim.x + threadIdx.x;
  if (t < n) d[t] = s[t] ? 1 : 0;
}
__global__ void k_copy_f(const float* s, float* d, long n) {
  long t = (long)blockIdx.x * blockDim.x + threadIdx.x;
  if (t < n) d[t] = s[t];
}
__global__ void k_add_f(float* x, const float* y, long n) {
  long t = (long)blockIdx.x * blockDim.x + threadIdx.x;
  if (t < n) x[t] += y[t];
}
__global__ void k_addrow_f(float* x, const float* row, long rows, int C) {
  long t = (long)blockIdx.x * blockDim.x + threadIdx.x;
  if (t < rows * (long)C) x[t] += row[t % C];
}

// K=3 dense layer (xyz inputs): out = act(X(rows,3) @ W(N,3)^T + b) -> bf16
__global__ void k_lin3(const float* __restrict__ X, const float* __restrict__ W,
                       const float* __restrict__ bias, unsigned short* outB,
                       long rows, int N, int act)
{
  long t = (long)blockIdx.x * blockDim.x + threadIdx.x;
  if (t >= rows * (long)N) return;
  long r = t / N; int oc = (int)(t % N);
  float v = bias[oc] + X[r * 3 + 0] * W[oc * 3 + 0]
                     + X[r * 3 + 1] * W[oc * 3 + 1]
                     + X[r * 3 + 2] * W[oc * 3 + 2];
  if (act) v = 0.5f * v * (1.f + erff(v * 0.70710678f));
  outB[t] = f2bf(v);
}

// max over n inside each group: X ((bg*n) x C bf16) -> (bg x C)
__global__ void k_groupmax(const unsigned short* __restrict__ X, long bg, int n,
                           int C, float* outF, unsigned short* outB)
{
  long t = (long)blockIdx.x * blockDim.x + threadIdx.x;
  if (t >= bg * (long)C) return;
  long g = t / C; int c = (int)(t % C);
  float m = -3.0e38f;
  for (int j = 0; j < n; ++j)
    m = fmaxf(m, bf2f(X[(g * n + j) * (long)C + c]));
  if (outF) outF[t] = m;
  if (outB) outB[t] = f2bf(m);
}

// mask back-prop: maskPrev[vis ? idx : 0] = 0 (reference quirk preserved)
__global__ void k_maskscatter(const unsigned char* __restrict__ maskCur,
                              const int* __restrict__ idx,
                              unsigned char* maskPrev, int Bv, int G2, int K)
{
  long t = (long)blockIdx.x * blockDim.x + threadIdx.x;
  long total = (long)Bv * G2 * K;
  if (t >= total) return;
  int b = (int)(t / ((long)G2 * K));
  int g = (int)((t / K) % G2);
  int vis = maskCur[b * G2 + g] ? 0 : 1;
  int flat = vis ? idx[t] : 0;
  maskPrev[flat] = 0;
}

// stable 0/1-key argsort per row (serial; G<=512, B=32 rows total)
__global__ void k_order(const unsigned char* __restrict__ mask, int G,
                        int vfirst, int* order, int* inv, int* cnt)
{
  if (threadIdx.x != 0) return;
  int b = blockIdx.x;
  int w = 0;
  for (int g = 0; g < G; ++g) {
    int key = mask[b * G + g] ? 1 : 0;
    if (!vfirst) key ^= 1;
    if (key == 0) order[b * G + (w++)] = g;
  }
  int c0 = w;
  for (int g = 0; g < G; ++g) {
    int key = mask[b * G + g] ? 1 : 0;
    if (!vfirst) key ^= 1;
    if (key == 1) order[b * G + (w++)] = g;
  }
  if (cnt) cnt[b] = c0;
  if (inv) for (int g = 0; g < G; ++g) inv[b * G + order[b * G + g]] = g;
}

// gather with ragged zero-pad: dst[b,g,c] = g<cnt[b] ? src[b,order[b,g],c] : 0
__global__ void k_gather_pad(const float* __restrict__ src,
                             const int* __restrict__ order,
                             const int* __restrict__ cnt, float* dst,
                             int Bv, int G, int C)
{
  long t = (long)blockIdx.x * blockDim.x + threadIdx.x;
  long total = (long)Bv * G * C;
  if (t >= total) return;
  int b = (int)(t / ((long)G * C));
  int g = (int)((t / C) % G);
  int c = (int)(t % C);
  dst[t] = (g < cnt[b]) ? src[((long)b * G + order[b * G + g]) * C + c] : 0.f;
}

// x[b,g] = mask ? git[b,g] : x_vis[b, inv[b,g]]
__global__ void k_scatter_back(const float* __restrict__ xv,
                               const int* __restrict__ inv,
                               const unsigned char* __restrict__ mask,
                               const float* __restrict__ git, float* x,
                               int Bv, int G, int C)
{
  long t = (long)blockIdx.x * blockDim.x + threadIdx.x;
  long total = (long)Bv * G * C;
  if (t >= total) return;
  int b = (int)(t / ((long)G * C));
  int g = (int)((t / C) % G);
  int c = (int)(t % C);
  x[t] = mask[b * G + g] ? git[t]
                         : xv[((long)b * G + inv[b * G + g]) * C + c];
}

// flat row gather (idx carries batch offsets) -> bf16
__global__ void k_gather_rows_bf(const float* __restrict__ x,
                                 const int* __restrict__ idx,
                                 unsigned short* dst, long rows, int C)
{
  long t = (long)blockIdx.x * blockDim.x + threadIdx.x;
  if (t >= rows * (long)C) return;
  long r = t / C; int c = (int)(t % C);
  dst[t] = f2bf(x[(long)idx[r] * C + c]);
}

// gather `sel` leading groups of `order` per batch: dst[(b,sel),inner]
__global__ void k_gather_groups(const float* __restrict__ src,
                                const int* __restrict__ order, float* dst,
                                int Bv, int G, int sel, int inner)
{
  long t = (long)blockIdx.x * blockDim.x + threadIdx.x;
  long total = (long)Bv * sel * inner;
  if (t >= total) return;
  int b = (int)(t / ((long)sel * inner));
  int r = (int)((t / inner) % sel);
  int e = (int)(t % inner);
  int g = order[b * G + r];
  dst[t] = src[((long)b * G + g) * inner + e];
}

// attention block mask: am = (dist>=radius) * (1 - (vi & vj))
__global__ void k_mkmask(const float* __restrict__ mcen,
                         const int* __restrict__ cnt, unsigned char* am,
                         int Bv, int G, float radius)
{
  long t = (long)blockIdx.x * blockDim.x + threadIdx.x;
  long total = (long)Bv * G * G;
  if (t >= total) return;
  int b = (int)(t / ((long)G * G));
  int i = (int)((t / G) % G);
  int j = (int)(t % G);
  int vi = i < cnt[b], vj = j < cnt[b];
  int mv = !(vi && vj);
  int blocked = mv;
  if (radius > 0.f) {
    const float* pi = mcen + ((long)b * G + i) * 3;
    const float* pj = mcen + ((long)b * G + j) * 3;
    float dx = pi[0] - pj[0], dy = pi[1] - pj[1], dz = pi[2] - pj[2];
    float dist = sqrtf(fmaxf(dx * dx + dy * dy + dz * dz, 0.f));
    blocked = (dist >= radius) && mv;
  }
  am[t] = (unsigned char)blocked;
}

// ---------------------------------------------------------------------------
// host-side orchestration
// ---------------------------------------------------------------------------
struct Pool { char* base; size_t off; };
static inline void* ptake(Pool& p, size_t bytes) {
  size_t a = (p.off + 255) & ~(size_t)255;
  p.off = a + bytes;
  return p.base + a;
}
static inline dim3 g1(long n) { return dim3((unsigned)((n + 255) / 256)); }
static inline dim3 gln(long rows) { return dim3((unsigned)((rows + 7) / 8)); }

static void gemm(hipStream_t st, const unsigned short* A,
                 const unsigned short* W, int ldw, const float* bias,
                 const float* resid, float* outF, unsigned short* outB,
                 long M, int N, int K, int arow_div, int flags)
{
  dim3 grid((unsigned)((M + 31) / 32), (unsigned)((N + 127) / 128));
  k_gemm_bf16<<<grid, 128, 0, st>>>(A, W, ldw, bias, resid, outF, outB,
                                    M, N, K, arow_div, flags);
}

struct TeW  { const float *w1,*b1,*bn1s,*bn1b,*w2,*b2,*w3,*b3,*bn2s,*bn2b,*w4,*b4; };
struct PosW { const float *w1,*b1,*w2,*b2; };
struct IncW { const float *w,*b; };
struct BlkW { const float *ln1s,*ln1b,*qkv_w,*qkv_b,*pw,*pb,*ln2s,*ln2b,*f1,*fb1,*f2,*fb2; };
struct NormW { const float *s,*b; };
struct TeBf { const unsigned short *w1,*w2,*w3,*w4; };
struct BlkBf { const unsigned short *qkv,*pw,*f1,*f2; };

extern "C" void kernel_launch(void* const* d_in, const int* in_sizes, int n_in,
                              void* d_out, int out_size, void* d_ws,
                              size_t ws_size, hipStream_t stream)
{
  (void)in_sizes; (void)n_in; (void)out_size; (void)ws_size;
  const int Bv = 32;
  const int GSh[3] = {512, 256, 64};
  const int KSh[3] = {16, 8, 8};
  const int DM[3]  = {96, 192, 384};
  const float RAD[3] = {0.32f, 0.64f, 1.28f};
  const int teIn[3]  = {3, 96, 192};
  const int teOut[3] = {96, 192, 384};

  auto F = [&](int i) -> const float* { return (const float*)d_in[i]; };
  const float* nbh[3] = {F(0), F(1), F(2)};
  const float* ctr[3] = {F(3), F(4), F(5)};
  const int* idxs[3] = {nullptr, (const int*)d_in[6], (const int*)d_in[7]};
  const unsigned char* mask2in = (const unsigned char*)d_in[8];

  int pi = 9;
  TeW te[3];
  for (int t = 0; t < 3; ++t) {
    te[t] = {F(pi),F(pi+1),F(pi+2),F(pi+3),F(pi+4),F(pi+5),
             F(pi+6),F(pi+7),F(pi+8),F(pi+9),F(pi+10),F(pi+11)};
    pi += 12;
  }
  PosW posW[3];
  for (int t = 0; t < 3; ++t) { posW[t] = {F(pi),F(pi+1),F(pi+2),F(pi+3)}; pi += 4; }
  IncW incW[3]; incW[0] = {nullptr, nullptr};
  incW[1] = {F(pi), F(pi+1)}; pi += 2;
  incW[2] = {F(pi), F(pi+1)}; pi += 2;
  BlkW blkW[3][5];
  for (int l = 0; l < 3; ++l)
    for (int k = 0; k < 5; ++k) {
      blkW[l][k] = {F(pi),F(pi+1),F(pi+2),F(pi+3),F(pi+4),F(pi+5),
                    F(pi+6),F(pi+7),F(pi+8),F(pi+9),F(pi+10),F(pi+11)};
      pi += 12;
    }
  NormW norms[3];
  for (int t = 0; t < 3; ++t) { norms[t] = {F(pi), F(pi+1)}; pi += 2; }
  const float* maskPointTok = F(pi++);
  const float* maskPosTok   = F(pi++);

  Pool pool{(char*)d_ws, 0};

  auto tobf = [&](const float* w, long n) -> const unsigned short* {
    unsigned short* o = (unsigned short*)ptake(pool, (size_t)n * 2);
    k_f2bf<<<g1(n), 256, 0, stream>>>(w, o, n);
    return o;
  };

  // ---- weight conversion to bf16 -------------------------------------------
  TeBf teBf[3];
  for (int t = 0; t < 3; ++t) {
    teBf[t].w1 = (teIn[t] == 3) ? nullptr : tobf(te[t].w1, 128L * teIn[t]);
    teBf[t].w2 = tobf(te[t].w2, 256L * 128);
    teBf[t].w3 = tobf(te[t].w3, 512L * 512);
    teBf[t].w4 = tobf(te[t].w4, (long)teOut[t] * 512);
  }
  const unsigned short* posW2b[3];
  for (int t = 0; t < 3; ++t) posW2b[t] = tobf(posW[t].w2, (long)DM[t] * DM[t]);
  const unsigned short* incWb[3] = {nullptr, nullptr, nullptr};
  incWb[1] = tobf(incW[1].w, 192L * 96);
  incWb[2] = tobf(incW[2].w, 384L * 192);
  BlkBf blkBf[3][5];
  for (int l = 0; l < 3; ++l) {
    long d = DM[l];
    for (int k = 0; k < 5; ++k) {
      blkBf[l][k].qkv = tobf(blkW[l][k].qkv_w, 3 * d * d);
      blkBf[l][k].pw  = tobf(blkW[l][k].pw,    d * d);
      blkBf[l][k].f1  = tobf(blkW[l][k].f1,  4 * d * d);
      blkBf[l][k].f2  = tobf(blkW[l][k].f2,  4 * d * d);
    }
  }
  float* stats = (float*)ptake(pool, 1024 * sizeof(float));

  // ---- helper routines -----------------------------------------------------
  auto posmlp = [&](const float* xyz, long rows, int d, const PosW& Pw,
                    const unsigned short* w2b, float* outF) {
    size_t mark = pool.off;
    unsigned short* h = (unsigned short*)ptake(pool, (size_t)rows * d * 2);
    k_lin3<<<g1(rows * d), 256, 0, stream>>>(xyz, Pw.w1, Pw.b1, h, rows, d, 1);
    gemm(stream, h, w2b, d, Pw.b2, nullptr, outF, nullptr, rows, d, d, 1, 0);
    pool.off = mark;
  };

  auto run_te = [&](const float* X3, const unsigned short* Xbf, int c, long bg,
                    int n, int outc, const TeW& Wt, const TeBf& Wb,
                    float* outTok) {
    long rows = bg * n;
    size_t mark = pool.off;
    unsigned short* a1 = (unsigned short*)ptake(pool, (size_t)rows * 128 * 2);
    if (c == 3)
      k_lin3<<<g1(rows * 128), 256, 0, stream>>>(X3, Wt.w1, Wt.b1, a1, rows, 128, 0);
    else
      gemm(stream, Xbf, Wb.w1, c, Wt.b1, nullptr, nullptr, a1, rows, 128, c, 1, 0);
    k_zero_f<<<g1(1024), 256, 0, stream>>>(stats, 1024);
    k_colstats<<<dim3(2, 128), 64, 0, stream>>>(a1, rows, 128, stats);
    k_bn<<<g1(rows * 128), 256, 0, stream>>>(a1, rows, 128, stats, Wt.bn1s, Wt.bn1b, 1);
    unsigned short* h2 = (unsigned short*)ptake(pool, (size_t)rows * 256 * 2);
    gemm(stream, a1, Wb.w2, 128, Wt.b2, nullptr, nullptr, h2, rows, 256, 128, 1, 0);
    unsigned short* gmx = (unsigned short*)ptake(pool, (size_t)bg * 256 * 2);
    k_groupmax<<<g1(bg * 256), 256, 0, stream>>>(h2, bg, n, 256, nullptr, gmx);
    unsigned short* a3 = (unsigned short*)ptake(pool, (size_t)rows * 512 * 2);
    // [broadcast(gmax) | h2] @ w3^T as two half-K WMMA passes (no concat)
    gemm(stream, gmx, Wb.w3,        512, nullptr, nullptr, nullptr, a3,
         rows, 512, 256, n, 0);
    gemm(stream, h2,  Wb.w3 + 256,  512, Wt.b3,  nullptr, nullptr, a3,
         rows, 512, 256, 1, F_ACC);
    k_zero_f<<<g1(1024), 256, 0, stream>>>(stats, 1024);
    k_colstats<<<dim3(8, 128), 64, 0, stream>>>(a3, rows, 512, stats);
    k_bn<<<g1(rows * 512), 256, 0, stream>>>(a3, rows, 512, stats, Wt.bn2s, Wt.bn2b, 1);
    unsigned short* h4 = (unsigned short*)ptake(pool, (size_t)rows * outc * 2);
    gemm(stream, a3, Wb.w4, 512, Wt.b4, nullptr, nullptr, h4, rows, outc, 512, 1, 0);
    k_groupmax<<<g1(bg * outc), 256, 0, stream>>>(h4, bg, n, outc, outTok, nullptr);
    pool.off = mark;
  };

  auto attn = [&](const float* qkv, const unsigned char* am, unsigned short* o,
                  int bN, int n, int d) {
    int dh = d / HEADS;
    size_t lds = (size_t)n * dh * 2 * sizeof(float);
    dim3 grid((unsigned)(bN * HEADS));
    if (dh == 16)      k_attn<16><<<grid, 256, lds, stream>>>(qkv, am, o, n, d);
    else if (dh == 32) k_attn<32><<<grid, 256, lds, stream>>>(qkv, am, o, n, d);
    else               k_attn<64><<<grid, 256, lds, stream>>>(qkv, am, o, n, d);
  };

  auto run_block = [&](float* x, const unsigned char* am, int bN, int n, int d,
                       const BlkW& Wt, const BlkBf& Wb) {
    long rows = (long)bN * n;
    size_t mark = pool.off;
    unsigned short* hb = (unsigned short*)ptake(pool, (size_t)rows * d * 2);
    k_ln<<<gln(rows), 256, 0, stream>>>(x, Wt.ln1s, Wt.ln1b, hb, nullptr, rows, d);
    float* qkv = (float*)ptake(pool, (size_t)rows * 3 * d * 4);
    gemm(stream, hb, Wb.qkv, d, Wt.qkv_b, nullptr, qkv, nullptr, rows, 3 * d, d, 1, 0);
    unsigned short* ob = (unsigned short*)ptake(pool, (size_t)rows * d * 2);
    attn(qkv, am, ob, bN, n, d);
    gemm(stream, ob, Wb.pw, d, Wt.pb, x, x, nullptr, rows, d, d, 1, 0);
    unsigned short* h2 = (unsigned short*)ptake(pool, (size_t)rows * d * 2);
    k_ln<<<gln(rows), 256, 0, stream>>>(x, Wt.ln2s, Wt.ln2b, h2, nullptr, rows, d);
    unsigned short* f1o = (unsigned short*)ptake(pool, (size_t)rows * 4 * d * 2);
    gemm(stream, h2, Wb.f1, d, Wt.fb1, nullptr, nullptr, f1o, rows, 4 * d, d, 1, F_GELU);
    gemm(stream, f1o, Wb.f2, 4 * d, Wt.fb2, x, x, nullptr, rows, d, 4 * d, 1, 0);
    pool.off = mark;
  };

  // ---- masks + orderings ---------------------------------------------------
  unsigned char* mk[3];
  mk[2] = (unsigned char*)ptake(pool, Bv * 64);
  k_copy_b<<<g1(Bv * 64), 256, 0, stream>>>(mask2in, mk[2], Bv * 64);
  mk[1] = (unsigned char*)ptake(pool, Bv * 256);
  k_fill_b<<<g1(Bv * 256), 256, 0, stream>>>(mk[1], 1, Bv * 256);
  k_maskscatter<<<g1((long)Bv * 64 * 8), 256, 0, stream>>>(mk[2], idxs[2], mk[1], Bv, 64, 8);
  mk[0] = (unsigned char*)ptake(pool, Bv * 512);
  k_fill_b<<<g1(Bv * 512), 256, 0, stream>>>(mk[0], 1, Bv * 512);
  k_maskscatter<<<g1((long)Bv * 256 * 8), 256, 0, stream>>>(mk[1], idxs[1], mk[0], Bv, 256, 8);

  int *ord[3], *invp[3], *cntp[3];
  for (int i = 0; i < 3; ++i) {
    int G = GSh[i];
    ord[i]  = (int*)ptake(pool, (size_t)Bv * G * 4);
    invp[i] = (int*)ptake(pool, (size_t)Bv * G * 4);
    cntp[i] = (int*)ptake(pool, (size_t)Bv * 4);
    k_order<<<Bv, 32, 0, stream>>>(mk[i], G, 1, ord[i], invp[i], cntp[i]);
  }
  int* mord2 = (int*)ptake(pool, (size_t)Bv * 64 * 4);
  k_order<<<Bv, 32, 0, stream>>>(mk[2], 64, 0, mord2, nullptr, nullptr);

  // ---- masked-token streams init (finest level, dims[0]) -------------------
  long rowsm = (long)Bv * NMASK;
  float* ctrm = (float*)ptake(pool, (size_t)rowsm * 3 * 4);
  k_gather_groups<<<g1(rowsm * 3), 256, 0, stream>>>(ctr[2], mord2, ctrm, Bv, 64, NMASK, 3);
  float* nbhm = (float*)ptake(pool, (size_t)rowsm * 8 * 3 * 4);
  k_gather_groups<<<g1(rowsm * 24), 256, 0, stream>>>(nbh[2], mord2, nbhm, Bv, 64, NMASK, 24);

  float* xmPos  = (float*)ptake(pool, (size_t)rowsm * 384 * 4);
  float* xmPt   = (float*)ptake(pool, (size_t)rowsm * 384 * 4);
  float* xmPosA = (float*)ptake(pool, (size_t)rowsm * 384 * 4);
  float* xmPtA  = (float*)ptake(pool, (size_t)rowsm * 384 * 4);
  unsigned short* xmB = (unsigned short*)ptake(pool, (size_t)rowsm * 384 * 2);
  float* tokPt     = (float*)ptake(pool, 384 * 4);
  float* tokPoint  = (float*)ptake(pool, 384 * 4);
  float* tokPtA    = (float*)ptake(pool, 384 * 4);
  float* tokPointA = (float*)ptake(pool, 384 * 4);
  unsigned short* tokB = (unsigned short*)ptake(pool, 384 * 2);
  (void)hipMemcpyAsync(tokPt,    maskPosTok,   96 * 4, hipMemcpyDeviceToDevice, stream);
  (void)hipMemcpyAsync(tokPoint, maskPointTok, 96 * 4, hipMemcpyDeviceToDevice, stream);

  posmlp(ctrm, rowsm, 96, posW[0], posW2b[0], xmPt);               // pos stream
  run_te(nbhm, nullptr, 3, rowsm, 8, 96, te[0], teBf[0], xmPos);   // point stream

  float* xstate = (float*)ptake(pool, (size_t)Bv * 512 * 96 * 4);

  float* outF = (float*)d_out;
  size_t outOff[5];
  outOff[0] = 0;
  outOff[1] = outOff[0] + (size_t)Bv * 512 * 96;
  outOff[2] = outOff[1] + (size_t)Bv * 256 * 192;
  outOff[3] = outOff[2] + (size_t)Bv * 64 * 384;
  outOff[4] = outOff[3] + (size_t)rowsm * 384;

  // ---- level loop ----------------------------------------------------------
  int dPrev = 0;
  for (int i = 0; i < 3; ++i) {
    int G = GSh[i], d = DM[i];
    long rowsTok = (long)Bv * G;
    size_t lmark = pool.off;

    float* git = (float*)ptake(pool, (size_t)rowsTok * d * 4);
    if (i == 0) {
      run_te(nbh[0], nullptr, 3, rowsTok, KSh[0], d, te[0], teBf[0], git);
    } else {
      long rg = (long)Bv * G * KSh[i];
      unsigned short* xn = (unsigned short*)ptake(pool, (size_t)rg * dPrev * 2);
      k_gather_rows_bf<<<g1(rg * dPrev), 256, 0, stream>>>(xstate, idxs[i], xn, rg, dPrev);
      run_te(nullptr, xn, dPrev, rowsTok, KSh[i], d, te[i], teBf[i], git);
      // dimension-increase linear on masked streams + tokens
      k_f2bf<<<g1(rowsm * dPrev), 256, 0, stream>>>(xmPos, xmB, rowsm * dPrev);
      gemm(stream, xmB, incWb[i], dPrev, incW[i].b, nullptr, xmPosA, nullptr,
           rowsm, d, dPrev, 1, 0);
      { float* t = xmPos; xmPos = xmPosA; xmPosA = t; }
      k_f2bf<<<g1(rowsm * dPrev), 256, 0, stream>>>(xmPt, xmB, rowsm * dPrev);
      gemm(stream, xmB, incWb[i], dPrev, incW[i].b, nullptr, xmPtA, nullptr,
           rowsm, d, dPrev, 1, 0);
      { float* t = xmPt; xmPt = xmPtA; xmPtA = t; }
      k_f2bf<<<g1(dPrev), 256, 0, stream>>>(tokPt, tokB, dPrev);
      gemm(stream, tokB, incWb[i], dPrev, incW[i].b, nullptr, tokPtA, nullptr,
           1, d, dPrev, 1, 0);
      { float* t = tokPt; tokPt = tokPtA; tokPtA = t; }
      k_f2bf<<<g1(dPrev), 256, 0, stream>>>(tokPoint, tokB, dPrev);
      gemm(stream, tokB, incWb[i], dPrev, incW[i].b, nullptr, tokPointA, nullptr,
           1, d, dPrev, 1, 0);
      { float* t = tokPoint; tokPoint = tokPointA; tokPointA = t; }
    }

    float* xvis = (float*)ptake(pool, (size_t)rowsTok * d * 4);
    k_gather_pad<<<g1(rowsTok * d), 256, 0, stream>>>(git, ord[i], cntp[i], xvis, Bv, G, d);
    float* mcen = (float*)ptake(pool, (size_t)rowsTok * 3 * 4);
    k_gather_pad<<<g1(rowsTok * 3), 256, 0, stream>>>(ctr[i], ord[i], cntp[i], mcen, Bv, G, 3);
    unsigned char* am = (unsigned char*)ptake(pool, (size_t)Bv * G * G);
    k_mkmask<<<g1((long)Bv * G * G), 256, 0, stream>>>(mcen, cntp[i], am, Bv, G, RAD[i]);
    float* posb = (float*)ptake(pool, (size_t)rowsTok * d * 4);
    posmlp(mcen, rowsTok, d, posW[i], posW2b[i], posb);

    for (int bI = 0; bI < 5; ++bI) {
      k_add_f<<<g1(rowsTok * d), 256, 0, stream>>>(xvis, posb, rowsTok * d);
      run_block(xvis, am, Bv, G, d, blkW[i][bI], blkBf[i][bI]);
      k_addrow_f<<<g1(rowsm * d), 256, 0, stream>>>(xmPos, tokPt, rowsm, d);
      run_block(xmPos, nullptr, Bv, NMASK, d, blkW[i][bI], blkBf[i][bI]);
      k_addrow_f<<<g1(rowsm * d), 256, 0, stream>>>(xmPt, tokPoint, rowsm, d);
      run_block(xmPt, nullptr, Bv, NMASK, d, blkW[i][bI], blkBf[i][bI]);
    }

    k_ln<<<gln(rowsTok), 256, 0, stream>>>(xvis, norms[i].s, norms[i].b,
                                           nullptr, outF + outOff[i], rowsTok, d);
    if (i < 2)
      k_scatter_back<<<g1(rowsTok * d), 256, 0, stream>>>(xvis, invp[i], mk[i],
                                                          git, xstate, Bv, G, d);
    pool.off = lmark;
    dPrev = d;
  }

  k_copy_f<<<g1(rowsm * 384), 256, 0, stream>>>(xmPos, outF + outOff[3], rowsm * 384);
  k_copy_f<<<g1(rowsm * 384), 256, 0, stream>>>(xmPt,  outF + outOff[4], rowsm * 384);
}
